// TopicDistQuant_25769803776029
// MI455X (gfx1250) — matrix operand order repaired and checked by reading, not compiled, not run on hardware
//
#include <hip/hip_runtime.h>
#include <hip/hip_bf16.h>

// MI455X / gfx1250, wave32. VQ codebook quantization fused kernel.
// Outputs in d_out (float): [0] loss, [1 .. N*D] quantized, [1+N*D ..] indices.

#define NROWS 131072
#define KCODES 512
#define DDIM 512

typedef __attribute__((ext_vector_type(16))) __bf16 v16bf;
typedef __attribute__((ext_vector_type(8)))  float  v8f;
typedef __attribute__((ext_vector_type(4)))  float  f32x4;

union BF16x16 { v16bf v; unsigned int u[8]; };

__device__ __forceinline__ unsigned int pack2_bf16(float lo, float hi) {
    unsigned int a = __float_as_uint(lo);
    unsigned int b = __float_as_uint(hi);
    // round-to-nearest-even f32 -> bf16
    a += 0x7FFFu + ((a >> 16) & 1u);
    b += 0x7FFFu + ((b >> 16) & 1u);
    return (a >> 16) | (b & 0xFFFF0000u);
}

// Kernel 0: codebook f32 -> bf16 (row-major K x D), per-row squared norms,
// and zero the scalar loss accumulator. One block per code row.
__global__ __launch_bounds__(256)
void vq_prep_kernel(const float* __restrict__ emb,
                    unsigned int* __restrict__ emb_bf,   // K*D/2 packed u32
                    float* __restrict__ enorm,           // K
                    float* __restrict__ d_out) {
    const int k = blockIdx.x;
    const int t = threadIdx.x;          // 256 threads, 2 columns each
    const float x0 = emb[k * DDIM + 2 * t + 0];
    const float x1 = emb[k * DDIM + 2 * t + 1];
    emb_bf[k * (DDIM / 2) + t] = pack2_bf16(x0, x1);

    float ss = x0 * x0 + x1 * x1;
#pragma unroll
    for (int s = 16; s >= 1; s >>= 1) ss += __shfl_xor(ss, s, 32);

    __shared__ float red[8];
    if ((t & 31) == 0) red[t >> 5] = ss;
    __syncthreads();
    if (t == 0) {
        float tot = 0.f;
#pragma unroll
        for (int i = 0; i < 8; ++i) tot += red[i];
        enorm[k] = tot;
        if (k == 0) d_out[0] = 0.f;     // loss accumulator
    }
}

// Main kernel: each wave owns 16 input rows. WMMA distance matmul + fused
// argmin + gather/quantized write + loss reduction.
__global__ __launch_bounds__(128)
void vq_main_kernel(const float* __restrict__ in,        // N x D f32
                    const float* __restrict__ emb,       // K x D f32
                    const unsigned int* __restrict__ emb_bf, // K x D bf16 (u32 pairs)
                    const float* __restrict__ enorm,     // K
                    float* __restrict__ d_out) {
    const int lane = threadIdx.x & 31;
    const int w    = threadIdx.x >> 5;            // wave in block (0..3)
    const int tile = blockIdx.x * 4 + w;          // 16-row tile id
    const int rowbase = tile << 4;
    const int m  = lane & 15;                     // row (A) / col (B,C) within tile
    const int hi = lane >> 4;                     // lane half

    const float* __restrict__ irow = in + (size_t)(rowbase + m) * DDIM;

    // ---- Load A operand: 16 rows x 512 cols bf16, ISA 16-bit A 16x32 layout.
    // lane half hi: halfs 0..7 = K[k0+8*hi .. +8), halfs 8..15 = K[k0+16+8*hi .. +8)
    BF16x16 a[16];
#pragma unroll
    for (int kb = 0; kb < 16; ++kb) {
        const float* p = irow + kb * 32 + hi * 8;
        const float4 f0 = *(const float4*)(p + 0);
        const float4 f1 = *(const float4*)(p + 4);
        const float4 f2 = *(const float4*)(p + 16);
        const float4 f3 = *(const float4*)(p + 20);
        a[kb].u[0] = pack2_bf16(f0.x, f0.y);
        a[kb].u[1] = pack2_bf16(f0.z, f0.w);
        a[kb].u[2] = pack2_bf16(f1.x, f1.y);
        a[kb].u[3] = pack2_bf16(f1.z, f1.w);
        a[kb].u[4] = pack2_bf16(f2.x, f2.y);
        a[kb].u[5] = pack2_bf16(f2.z, f2.w);
        a[kb].u[6] = pack2_bf16(f3.x, f3.y);
        a[kb].u[7] = pack2_bf16(f3.z, f3.w);
    }

    // Per-lane running best: lane sees only column (lane&15) of every tile,
    // so the per-tile step is a plain compare-select; the 16-way cross-lane
    // argmin reduction happens ONCE after the sweep (keeps VALU off the
    // WMMA-coexec path in the hot loop).
    float best[8];
    int   bidx[8];
#pragma unroll
    for (int r = 0; r < 8; ++r) { best[r] = 3.4e38f; bidx[r] = 0x7fffffff; }

    // ---- Sweep 32 column tiles of 16 codes each.
    for (int t = 0; t < 32; ++t) {
        const int codebase = t << 4;
        const int col = codebase + m;
        const float en = enorm[col];              // per-column ||e||^2
        // B operand base: code row (codebase+m), halfs k0 + 8*hi .. +8 and k0 + 16 + 8*hi .. +8
        const unsigned int* bp = emb_bf + (size_t)col * (DDIM / 2) + hi * 4;

        v8f acc = {0.f, 0.f, 0.f, 0.f, 0.f, 0.f, 0.f, 0.f};
#pragma unroll
        for (int kb = 0; kb < 16; ++kb) {
            BF16x16 b;
            const uint4 q0 = *(const uint4*)(bp + kb * 16 + 0);
            const uint4 q1 = *(const uint4*)(bp + kb * 16 + 8);
            b.u[0] = q0.x; b.u[1] = q0.y; b.u[2] = q0.z; b.u[3] = q0.w;
            b.u[4] = q1.x; b.u[5] = q1.y; b.u[6] = q1.z; b.u[7] = q1.w;
            acc = __builtin_amdgcn_wmma_f32_16x16x32_bf16(
                false, a[kb].v, false, b.v, (short)0, acc, false, false);
        }

        // dist(row, col) = ||e||^2 - 2 * x.e  (||x||^2 constant per row -> dropped)
        // Strict < keeps the earliest (lowest-index) tile on ties.
#pragma unroll
        for (int r = 0; r < 8; ++r) {
            const float d = __builtin_fmaf(-2.0f, acc[r], en);
            if (d < best[r]) { best[r] = d; bidx[r] = col; }
        }
    }

    // ---- One cross-lane min+argmin per row (stays within 16-lane half).
#pragma unroll
    for (int r = 0; r < 8; ++r) {
        float d  = best[r];
        int   ix = bidx[r];
#pragma unroll
        for (int s = 8; s >= 1; s >>= 1) {
            const float od = __shfl_xor(d, s, 32);
            const int   oi = __shfl_xor(ix, s, 32);
            if (od < d || (od == d && oi < ix)) { d = od; ix = oi; }
        }
        best[r] = d;
        bidx[r] = ix;
    }

    // ---- Exchange winning indices through LDS (row r+8*hi lives in half hi).
    __shared__ int sIdx[4][16];
    if (m == 0) {
#pragma unroll
        for (int r = 0; r < 8; ++r) sIdx[w][r + 8 * hi] = bidx[r];
    }
    __syncthreads();

    // ---- Epilogue: gather code row, write quantized (non-temporal: streamed,
    // never re-read -> keep L2 for the hot codebook), accumulate loss.
    const int code = sIdx[w][m];
    const float* __restrict__ erow = emb + (size_t)code * DDIM;
    float* __restrict__ qrow = d_out + 1 + (size_t)(rowbase + m) * DDIM;
    const int c0 = hi * (DDIM / 2);               // each half-lane does 256 cols
    float lsum = 0.f;
#pragma unroll 4
    for (int c = 0; c < DDIM / 2; c += 4) {
        const f32x4 e = *(const f32x4*)(erow + c0 + c);
        const f32x4 x = __builtin_nontemporal_load((const f32x4*)(irow + c0 + c));
        __builtin_nontemporal_store(e, (f32x4*)(qrow + c0 + c));
        const f32x4 dv = e - x;
        lsum += dv.x * dv.x + dv.y * dv.y + dv.z * dv.z + dv.w * dv.w;
    }

    if (hi == 0) {
        d_out[1 + (size_t)NROWS * DDIM + rowbase + m] = (float)code;
    }

#pragma unroll
    for (int s = 16; s >= 1; s >>= 1) lsum += __shfl_xor(lsum, s, 32);
    if (lane == 0) {
        // loss = q_latent + 0.1 * e_latent = 1.1 * mean(sum_d sq_err)
        atomicAdd(d_out, lsum * (1.1f / (float)NROWS));
    }
}

extern "C" void kernel_launch(void* const* d_in, const int* in_sizes, int n_in,
                              void* d_out, int out_size, void* d_ws, size_t ws_size,
                              hipStream_t stream) {
    const float* in  = (const float*)d_in[0];   // [N, D] f32
    const float* emb = (const float*)d_in[1];   // [K, D] f32

    unsigned int* emb_bf = (unsigned int*)d_ws;                         // 512 KB
    float* enorm = (float*)((char*)d_ws + (size_t)KCODES * DDIM * 2);   // 2 KB
    float* out = (float*)d_out;

    vq_prep_kernel<<<KCODES, 256, 0, stream>>>(emb, emb_bf, enorm, out);
    vq_main_kernel<<<NROWS / 64, 128, 0, stream>>>(in, emb, emb_bf, enorm, out);
}
